// OldRouterModel_45621142618347
// MI455X (gfx1250) — matrix operand order repaired and checked by reading, not compiled, not run on hardware
//
#include <hip/hip_runtime.h>
#include <hip/hip_bf16.h>
#include <math.h>

// ---------------- problem constants (from reference) ----------------
#define NN 2048
#define EE 4096
#define DD 128
#define HH 4
#define DH 32
#define CC 16
#define NODE_K 1024
#define EDGE_K 2048
#define NEGV (-1.0e9f)

typedef __attribute__((ext_vector_type(16))) __bf16 v16bf;
typedef __attribute__((ext_vector_type(8)))  float  v8f;

// ---------------- helpers ----------------
__device__ __forceinline__ unsigned fkey(float f) {
    unsigned u = __float_as_uint(f);
    return (u & 0x80000000u) ? ~u : (u | 0x80000000u);
}

// A fragment, 16x32 bf16, row-major source, lda in elements.
// lanes 0-15: row = m0+lane, K = k0+{0..7, 16..23}
// lanes 16-31: row = m0+lane-16, K = k0+{8..15, 24..31}
__device__ __forceinline__ v16bf load_a_frag(const __bf16* A, int lda, int m0, int k0) {
    int lane = threadIdx.x & 31;
    int half = lane >> 4, r = lane & 15;
    const __bf16* p = A + (size_t)(m0 + r) * lda + k0 + (half ? 8 : 0);
    union { uint4 u[2]; v16bf v; } u;
    u.u[0] = *reinterpret_cast<const uint4*>(p);
    u.u[1] = *reinterpret_cast<const uint4*>(p + 16);
    return u.v;
}

// B fragment, K=32 x N=16 bf16, sourced from *transposed* storage Bt[N][K] (ldb = row len).
// lanes 0-15: col = n0+lane, K = k0+{0..7, 16..23} ; lanes 16-31: col = n0+lane-16, K = k0+{8..15, 24..31}
__device__ __forceinline__ v16bf load_bT_frag(const __bf16* Bt, int ldb, int n0, int k0) {
    int lane = threadIdx.x & 31;
    int half = lane >> 4, c = lane & 15;
    const __bf16* p = Bt + (size_t)(n0 + c) * ldb + k0 + (half ? 8 : 0);
    union { uint4 u[2]; v16bf v; } u;
    u.u[0] = *reinterpret_cast<const uint4*>(p);
    u.u[1] = *reinterpret_cast<const uint4*>(p + 16);
    return u.v;
}

__device__ __forceinline__ v8f wmma_bf16(v16bf a, v16bf b, v8f c) {
    return __builtin_amdgcn_wmma_f32_16x16x32_bf16(false, a, false, b, (short)0, c, false, false);
}

__device__ __forceinline__ v8f wmma_acc(const __bf16* A, int lda, const __bf16* Bt, int ldb,
                                        int m0, int n0, int K, v8f acc) {
    #pragma unroll
    for (int k0 = 0; k0 < K; k0 += 32) {
        v16bf a = load_a_frag(A, lda, m0, k0);
        v16bf b = load_bT_frag(Bt, ldb, n0, k0);
        acc = wmma_bf16(a, b, acc);
    }
    return acc;
}

__device__ __forceinline__ void store_c_bf16(__bf16* C, int ldc, int m0, int n0, v8f acc) {
    int lane = threadIdx.x & 31;
    int half = lane >> 4, c = lane & 15;
    int rb = m0 + (half ? 8 : 0);
    #pragma unroll
    for (int r = 0; r < 8; ++r)
        C[(size_t)(rb + r) * ldc + n0 + c] = (__bf16)acc[r];
}

__device__ __forceinline__ float rowmax16(float x) {
    #pragma unroll
    for (int m = 1; m < 16; m <<= 1) x = fmaxf(x, __shfl_xor(x, m, 32));
    return x;
}
__device__ __forceinline__ float rowsum16(float x) {
    #pragma unroll
    for (int m = 1; m < 16; m <<= 1) x += __shfl_xor(x, m, 32);
    return x;
}

// ---------------- kernel 1: router scores (one wave per row) ----------------
__global__ __launch_bounds__(256) void score_kernel(const float* __restrict__ nf,
                                                    const float* __restrict__ ef,
                                                    const float* __restrict__ wrn,
                                                    const float* __restrict__ wre,
                                                    float* __restrict__ ns,
                                                    float* __restrict__ es) {
    int gid = blockIdx.x * blockDim.x + threadIdx.x;
    int row = gid >> 5, lane = gid & 31;
    const float* f; const float* wv; float* out;
    if (row < NN)           { f = nf + (size_t)row * DD;        wv = wrn; out = ns + row; }
    else if (row < NN + EE) { int r = row - NN;
                              f = ef + (size_t)r * DD;          wv = wre; out = es + r; }
    else return;
    float s = 0.f;
    for (int i = lane; i < DD; i += 32) s += f[i] * wv[i];
    #pragma unroll
    for (int m = 16; m; m >>= 1) s += __shfl_xor(s, m, 32);
    if (lane == 0) *out = s;
}

// ---------------- kernel 2: exact top-K threshold via radix descend ----------------
__global__ __launch_bounds__(256) void topk_kernel(const float* __restrict__ ns,
                                                   const float* __restrict__ es,
                                                   unsigned* __restrict__ thr) {
    const float* v = blockIdx.x ? es : ns;
    int n          = blockIdx.x ? EE : NN;
    unsigned K     = blockIdx.x ? EDGE_K : NODE_K;
    __shared__ unsigned cnt;
    unsigned prefix = 0;
    for (int bit = 31; bit >= 0; --bit) {
        unsigned test = prefix | (1u << bit);
        if (threadIdx.x == 0) cnt = 0;
        __syncthreads();
        unsigned c = 0;
        for (int i = threadIdx.x; i < n; i += 256)
            if (fkey(v[i]) >= test) ++c;
        atomicAdd(&cnt, c);
        __syncthreads();
        if (cnt >= K) prefix = test;   // max u s.t. count(key>=u) >= K  == Kth largest
        __syncthreads();
    }
    if (threadIdx.x == 0) thr[blockIdx.x] = prefix;
}

// ---------------- kernel 3: weights -> transposed bf16 ----------------
__global__ void convw_kernel(const float* we, const float* wn, const float* wq,
                             const float* wk, const float* wv, const float* wo,
                             const float* w1, const float* w2,
                             __bf16* weT, __bf16* wnT, __bf16* wqT, __bf16* wkT,
                             __bf16* wvT, __bf16* woT, __bf16* w1T, __bf16* w2T) {
    const float* W; __bf16* O; int Ncols;
    switch (blockIdx.z) {
        case 0: W = we; O = weT; Ncols = DD; break;
        case 1: W = wn; O = wnT; Ncols = DD; break;
        case 2: W = wq; O = wqT; Ncols = DD; break;
        case 3: W = wk; O = wkT; Ncols = DD; break;
        case 4: W = wv; O = wvT; Ncols = DD; break;
        case 5: W = wo; O = woT; Ncols = DD; break;
        case 6: W = w1; O = w1T; Ncols = DD; break;
        default: W = w2; O = w2T; Ncols = CC; break;
    }
    int idx = blockIdx.x * blockDim.x + threadIdx.x;
    if (idx < DD * Ncols) {
        int k = idx / Ncols, n = idx % Ncols;
        O[(size_t)n * DD + k] = (__bf16)W[idx];   // Bt[N][K]
    }
}

// ---------------- kernel 4: masked ef + endpoint node sum -> bf16 ----------------
__global__ __launch_bounds__(256) void build_kernel(const float* __restrict__ ef,
                                                    const float* __restrict__ nf,
                                                    const int* __restrict__ src,
                                                    const int* __restrict__ dst,
                                                    const float* __restrict__ ns,
                                                    const float* __restrict__ es,
                                                    const unsigned* __restrict__ thr,
                                                    __bf16* __restrict__ efb,
                                                    __bf16* __restrict__ nsb) {
    int idx = blockIdx.x * 256 + threadIdx.x;     // e*128 + d
    int e = idx >> 7, d = idx & 127;
    unsigned thrN = thr[0], thrE = thr[1];
    int s = src[e], t = dst[e];
    bool m = (fkey(es[e]) >= thrE) && (fkey(ns[s]) >= thrN) && (fkey(ns[t]) >= thrN);
    efb[idx] = m ? (__bf16)ef[idx] : (__bf16)0.0f;
    nsb[idx] = (__bf16)(nf[(size_t)s * DD + d] + nf[(size_t)t * DD + d]);
}

// ---------------- kernel 5: h = ef@We + nsum@Wn ----------------
__global__ __launch_bounds__(256) void gemm_h_kernel(const __bf16* efb, const __bf16* nsb,
                                                     const __bf16* weT, const __bf16* wnT,
                                                     __bf16* h) {
    int w  = threadIdx.x >> 5;
    int m0 = blockIdx.x * 32 + (w & 1) * 16;
    int n0 = blockIdx.y * 64 + (w >> 1) * 16;
    v8f acc = {};
    acc = wmma_acc(efb, DD, weT, DD, m0, n0, DD, acc);
    acc = wmma_acc(nsb, DD, wnT, DD, m0, n0, DD, acc);
    store_c_bf16(h, DD, m0, n0, acc);
}

// ---------------- kernel 6: q/k row-major, v stored transposed [DD][EE] ----------------
__global__ __launch_bounds__(256) void gemm_qkv_kernel(const __bf16* h,
                                                       const __bf16* wqT, const __bf16* wkT,
                                                       const __bf16* wvT,
                                                       __bf16* q, __bf16* k, __bf16* vT) {
    int w  = threadIdx.x >> 5;
    int m0 = blockIdx.x * 32 + (w & 1) * 16;
    int n0 = blockIdx.y * 64 + (w >> 1) * 16;
    const __bf16* Bt = (blockIdx.z == 0) ? wqT : (blockIdx.z == 1 ? wkT : wvT);
    v8f acc = {};
    acc = wmma_acc(h, DD, Bt, DD, m0, n0, DD, acc);
    if (blockIdx.z == 2) {
        int lane = threadIdx.x & 31, half = lane >> 4, c = lane & 15;
        int rb = m0 + (half ? 8 : 0);
        #pragma unroll
        for (int r = 0; r < 8; ++r)
            vT[(size_t)(n0 + c) * EE + rb + r] = (__bf16)acc[r];
    } else {
        store_c_bf16(blockIdx.z ? k : q, DD, m0, n0, acc);
    }
}

// ---------------- kernel 7: flash attention over edges with adjacency mask ----------------
__global__ __launch_bounds__(128) void attn_kernel(const __bf16* __restrict__ qb,
                                                   const __bf16* __restrict__ kb,
                                                   const __bf16* __restrict__ vT,
                                                   const int* __restrict__ src,
                                                   const int* __restrict__ dst,
                                                   __bf16* __restrict__ ao) {
    __shared__ int qsrc[16], qdst[16];
    __shared__ __align__(16) __bf16 plds[4][16][32];
    int tid  = threadIdx.x;
    int w    = tid >> 5;                 // head
    int lane = tid & 31, half = lane >> 4, c = lane & 15;
    int m0   = blockIdx.x * 16;
    if (tid < 16) { qsrc[tid] = src[m0 + tid]; qdst[tid] = dst[m0 + tid]; }
    __syncthreads();

    v16bf qa = load_a_frag(qb, DD, m0, DH * w);    // [16 q, 32 dh] loaded once
    v8f acc0 = {}, acc1 = {};
    float m_run[8], l_run[8];
    #pragma unroll
    for (int r = 0; r < 8; ++r) { m_run[r] = -3.0e38f; l_run[r] = 0.f; }
    const float scl = 0.17677669529663687f;        // 1/sqrt(32)

    for (int kblk = 0; kblk < EE / 32; ++kblk) {
        int kt0 = kblk * 32;
        int kid0 = kt0 + c, kid1 = kt0 + 16 + c;
        int ks0 = src[kid0], kd0 = dst[kid0];
        int ks1 = src[kid1], kd1 = dst[kid1];

        v16bf kf0 = load_bT_frag(kb + DH * w, DD, kt0,      0);
        v16bf kf1 = load_bT_frag(kb + DH * w, DD, kt0 + 16, 0);
        v8f z = {};
        v8f s0 = wmma_bf16(qa, kf0, z);
        v8f s1 = wmma_bf16(qa, kf1, z);
        if (kblk + 1 < EE / 32)
            __builtin_prefetch(kb + (size_t)(kt0 + 32 + c) * DD + DH * w, 0, 1);

        float sc[8];
        #pragma unroll
        for (int r = 0; r < 8; ++r) {
            int qrow = r + (half ? 8 : 0);
            int qs = qsrc[qrow], qd = qdst[qrow];
            float v0 = ((qs == ks0) | (qs == kd0) | (qd == ks0) | (qd == kd0)) ? s0[r] * scl : NEGV;
            float v1 = ((qs == ks1) | (qs == kd1) | (qd == ks1) | (qd == kd1)) ? s1[r] * scl : NEGV;
            float mx = rowmax16(fmaxf(v0, v1));
            float mn = fmaxf(m_run[r], mx);
            float p0 = __expf(v0 - mn), p1 = __expf(v1 - mn);
            float rs = rowsum16(p0 + p1);
            float scale = __expf(m_run[r] - mn);
            l_run[r] = l_run[r] * scale + rs;
            m_run[r] = mn;
            sc[r] = scale;
            plds[w][qrow][c]      = (__bf16)p0;
            plds[w][qrow][16 + c] = (__bf16)p1;
        }
        __syncthreads();
        #pragma unroll
        for (int r = 0; r < 8; ++r) { acc0[r] *= sc[r]; acc1[r] *= sc[r]; }

        v16bf pa  = load_a_frag(&plds[0][0][0] + (size_t)w * 16 * 32, 32, 0, 0);
        v16bf vb0 = load_bT_frag(vT + (size_t)(DH * w) * EE, EE, 0,  kt0);
        v16bf vb1 = load_bT_frag(vT + (size_t)(DH * w) * EE, EE, 16, kt0);
        acc0 = wmma_bf16(pa, vb0, acc0);
        acc1 = wmma_bf16(pa, vb1, acc1);
        __syncthreads();
    }

    #pragma unroll
    for (int r = 0; r < 8; ++r) {
        int row = m0 + r + (half ? 8 : 0);
        float inv = 1.0f / l_run[r];
        ao[(size_t)row * DD + DH * w + c]      = (__bf16)(acc0[r] * inv);
        ao[(size_t)row * DD + DH * w + 16 + c] = (__bf16)(acc1[r] * inv);
    }
}

// ---------------- kernel 8: o = attn_out @ Wo ----------------
__global__ __launch_bounds__(256) void gemm_o_kernel(const __bf16* ao, const __bf16* woT,
                                                     __bf16* o) {
    int w  = threadIdx.x >> 5;
    int m0 = blockIdx.x * 32 + (w & 1) * 16;
    int n0 = blockIdx.y * 64 + (w >> 1) * 16;
    v8f acc = {};
    acc = wmma_acc(ao, DD, woT, DD, m0, n0, DD, acc);
    store_c_bf16(o, DD, m0, n0, acc);
}

// ---------------- kernel 9: x = gelu(o @ W1 + b1) ----------------
__global__ __launch_bounds__(256) void gemm_gelu_kernel(const __bf16* o, const __bf16* w1T,
                                                        const float* __restrict__ b1,
                                                        __bf16* x) {
    int w  = threadIdx.x >> 5;
    int m0 = blockIdx.x * 32 + (w & 1) * 16;
    int n0 = blockIdx.y * 64 + (w >> 1) * 16;
    v8f acc = {};
    acc = wmma_acc(o, DD, w1T, DD, m0, n0, DD, acc);
    int lane = threadIdx.x & 31, half = lane >> 4, c = lane & 15;
    int rb = m0 + (half ? 8 : 0);
    float b = b1[n0 + c];
    #pragma unroll
    for (int r = 0; r < 8; ++r) {
        float xv = acc[r] + b;
        float t  = 0.7978845608028654f * (xv + 0.044715f * xv * xv * xv);
        x[(size_t)(rb + r) * DD + n0 + c] = (__bf16)(0.5f * xv * (1.f + tanhf(t)));
    }
}

// ---------------- kernel 10: logits = x @ W2 + b2 (f32 out) ----------------
__global__ __launch_bounds__(256) void gemm_logits_kernel(const __bf16* x, const __bf16* w2T,
                                                          const float* __restrict__ b2,
                                                          float* __restrict__ out) {
    int w  = threadIdx.x >> 5;
    int m0 = blockIdx.x * 128 + w * 16;
    v8f acc = {};
    acc = wmma_acc(x, DD, w2T, DD, m0, 0, DD, acc);
    int lane = threadIdx.x & 31, half = lane >> 4, c = lane & 15;
    int rb = m0 + (half ? 8 : 0);
    float b = b2[c];
    #pragma unroll
    for (int r = 0; r < 8; ++r)
        out[(size_t)(rb + r) * CC + c] = acc[r] + b;
}

// ---------------- launch ----------------
extern "C" void kernel_launch(void* const* d_in, const int* in_sizes, int n_in,
                              void* d_out, int out_size, void* d_ws, size_t ws_size,
                              hipStream_t stream) {
    const float* nf  = (const float*)d_in[0];
    const float* ef  = (const float*)d_in[1];
    const int*   ei  = (const int*)d_in[2];
    const float* wrn = (const float*)d_in[3];
    const float* wre = (const float*)d_in[4];
    const float* w_e = (const float*)d_in[5];
    const float* w_n = (const float*)d_in[6];
    const float* w_q = (const float*)d_in[7];
    const float* w_k = (const float*)d_in[8];
    const float* w_v = (const float*)d_in[9];
    const float* w_o = (const float*)d_in[10];
    const float* w1  = (const float*)d_in[11];
    const float* b1  = (const float*)d_in[12];
    const float* w2  = (const float*)d_in[13];
    const float* b2  = (const float*)d_in[14];
    const int* src = ei;
    const int* dst = ei + EE;
    float* out = (float*)d_out;

    char* ws = (char*)d_ws;
    size_t off = 0;
    auto alloc = [&](size_t bytes) -> void* {
        void* p = ws + off;
        off += (bytes + 255) & ~(size_t)255;
        return p;
    };
    float*    ns   = (float*)alloc(NN * 4);
    float*    es   = (float*)alloc(EE * 4);
    unsigned* thr  = (unsigned*)alloc(256);
    __bf16* weT = (__bf16*)alloc(DD * DD * 2);
    __bf16* wnT = (__bf16*)alloc(DD * DD * 2);
    __bf16* wqT = (__bf16*)alloc(DD * DD * 2);
    __bf16* wkT = (__bf16*)alloc(DD * DD * 2);
    __bf16* wvT = (__bf16*)alloc(DD * DD * 2);
    __bf16* woT = (__bf16*)alloc(DD * DD * 2);
    __bf16* w1T = (__bf16*)alloc(DD * DD * 2);
    __bf16* w2T = (__bf16*)alloc(CC * DD * 2);
    __bf16* efb  = (__bf16*)alloc((size_t)EE * DD * 2);
    __bf16* nsb  = (__bf16*)alloc((size_t)EE * DD * 2);
    __bf16* hbuf = (__bf16*)alloc((size_t)EE * DD * 2);
    __bf16* qbuf = (__bf16*)alloc((size_t)EE * DD * 2);
    __bf16* kbuf = (__bf16*)alloc((size_t)EE * DD * 2);
    __bf16* vTb  = (__bf16*)alloc((size_t)DD * EE * 2);
    __bf16* aob  = (__bf16*)alloc((size_t)EE * DD * 2);
    __bf16* obuf = (__bf16*)alloc((size_t)EE * DD * 2);
    __bf16* xbuf = (__bf16*)alloc((size_t)EE * DD * 2);

    // 1. router scores: one wave per row, (N+E) rows
    score_kernel<<<dim3(((NN + EE) * 32) / 256), dim3(256), 0, stream>>>(nf, ef, wrn, wre, ns, es);
    // 2. exact top-K thresholds (block 0: nodes, block 1: edges)
    topk_kernel<<<dim3(2), dim3(256), 0, stream>>>(ns, es, thr);
    // 3. weights -> transposed bf16
    convw_kernel<<<dim3(64, 1, 8), dim3(256), 0, stream>>>(w_e, w_n, w_q, w_k, w_v, w_o, w1, w2,
                                                           weT, wnT, wqT, wkT, wvT, woT, w1T, w2T);
    // 4. masked edge feats + endpoint node sum
    build_kernel<<<dim3((EE * DD) / 256), dim3(256), 0, stream>>>(ef, nf, src, dst, ns, es, thr,
                                                                  efb, nsb);
    // 5. h
    gemm_h_kernel<<<dim3(EE / 32, DD / 64), dim3(256), 0, stream>>>(efb, nsb, weT, wnT, hbuf);
    // 6. q, k, vT
    gemm_qkv_kernel<<<dim3(EE / 32, DD / 64, 3), dim3(256), 0, stream>>>(hbuf, wqT, wkT, wvT,
                                                                         qbuf, kbuf, vTb);
    // 7. flash attention (16-query tile per block, 1 wave per head)
    attn_kernel<<<dim3(EE / 16), dim3(128), 0, stream>>>(qbuf, kbuf, vTb, src, dst, aob);
    // 8. output projection
    gemm_o_kernel<<<dim3(EE / 32, DD / 64), dim3(256), 0, stream>>>(aob, woT, obuf);
    // 9. MLP layer 1 + GELU
    gemm_gelu_kernel<<<dim3(EE / 32, DD / 64), dim3(256), 0, stream>>>(obuf, w1T, b1, xbuf);
    // 10. logits (f32)
    gemm_logits_kernel<<<dim3(EE / 128), dim3(256), 0, stream>>>(xbuf, w2T, b2, out);
}
